// LiGRU_54760833024075
// MI455X (gfx1250) — matrix operand order repaired and checked by reading, not compiled, not run on hardware
//
#include <hip/hip_runtime.h>
#include <hip/hip_bf16.h>

typedef __bf16 bf16;
typedef __attribute__((ext_vector_type(16))) __bf16 v16bf;
typedef __attribute__((ext_vector_type(8)))  float   v8f;

struct Frag16 { uint4 lo, hi; };   // 32 bytes = one lane's 16 bf16 of an A/B fragment
static_assert(sizeof(Frag16) == sizeof(v16bf), "frag size mismatch");

#define NWG      32     // workgroups in persistent recurrence kernel
#define TSTEPS   500
#define BATCH2   32     // 2*B (fwd+bwd stacked)
#define HDIM     1024
#define GDIM     2048   // 2*H
#define ROWS_TOT 16000  // 2*B*T

__device__ inline v8f wmma_bf16(v16bf a, v16bf b, v8f c) {
  // D = A(16x32) * B(32x16) + C, fp32 accumulate
  return __builtin_amdgcn_wmma_f32_16x16x32_bf16(false, a, false, b, (short)0, c,
                                                 false, false);
}

// ---------------- fragment builders (fp32 memory -> bf16 fragments) -------------
// A (16x32 bf16): lane L: M = L&15, K-base = (L>>4)*8;
// VGPR pairs hold K = base + {0,2,4,6,16,18,20,22} (+1 in hi half).
__device__ inline v16bf load_A_f32(const float* rowp, int k0, int lane) {
  int kb = (lane >> 4) << 3;
  v16bf a;
#pragma unroll
  for (int v = 0; v < 8; ++v) {
    int k = k0 + kb + ((v < 4) ? (2 * v) : (8 + 2 * v));
    a[2 * v]     = (bf16)rowp[k];
    a[2 * v + 1] = (bf16)rowp[k + 1];
  }
  return a;
}

// B (32x16 bf16) for D = A * Wt^T:  B[k][n] = Wt[(n0+n)*ldb + k0 + k].
// lane L: N = L&15, K-base = (L>>4)*8; element h holds K = base + h (h<8), base + 8 + h (h>=8).
__device__ inline v16bf load_B_f32(const float* Wt, int ldb, int n0, int k0, int lane) {
  const float* p = Wt + (size_t)(n0 + (lane & 15)) * ldb + k0 + ((lane >> 4) << 3);
  v16bf b;
#pragma unroll
  for (int h = 0; h < 16; ++h) b[h] = (bf16)p[(h < 8) ? h : (h + 8)];
  return b;
}

// ---------------- pack U (2048x1024 fp32) into B-fragment layout (bf16) --------
// Upk[nt (0..127)][ks (0..31)][lane (0..31)] : one Frag16 per lane.
__global__ __launch_bounds__(256) void pack_u(const float* __restrict__ U,
                                              Frag16* __restrict__ Upk) {
  int id = blockIdx.x * 256 + threadIdx.x;          // 128*32*32 = 131072 total
  int lane = id & 31;
  int ks   = (id >> 5) & 31;
  int nt   = id >> 10;
  v16bf b = load_B_f32(U, HDIM, nt * 16, ks * 32, lane);
  Upk[id] = __builtin_bit_cast(Frag16, b);
}

// ---------------- pre-activation GEMM: w = x2 @ W^T  (rows 16000, cols 2048) ---
// x2 row r: b2=r/T, t=r%T; b2<16 -> x[b2,t], else x[b2-16, T-1-t] (time reversed).
__global__ __launch_bounds__(256) void pregemm(const float* __restrict__ X, int K,
                                               const float* __restrict__ W,
                                               bf16* __restrict__ wout) {
  const int tid = threadIdx.x, w = tid >> 5, lane = tid & 31;
  const int wr = w >> 2, wc = w & 3;                // 2x4 wave grid
  const int row0 = blockIdx.y * 64 + wr * 32;      // WG tile 64x128, wave tile 32x32
  const int col0 = blockIdx.x * 128 + wc * 32;
  const int lm = lane & 15;

  const float* arow[2];
#pragma unroll
  for (int mt = 0; mt < 2; ++mt) {
    int r  = row0 + mt * 16 + lm;
    int b2 = r / TSTEPS, t = r % TSTEPS;
    int src = (b2 < 16) ? (b2 * TSTEPS + t)
                        : ((b2 - 16) * TSTEPS + (TSTEPS - 1 - t));
    arow[mt] = X + (size_t)src * K;
  }

  v8f acc[2][2] = {};
  for (int k0 = 0; k0 < K; k0 += 32) {
    v16bf a0 = load_A_f32(arow[0], k0, lane);
    v16bf a1 = load_A_f32(arow[1], k0, lane);
    v16bf b0 = load_B_f32(W, K, col0,      k0, lane);
    v16bf b1 = load_B_f32(W, K, col0 + 16, k0, lane);
    acc[0][0] = wmma_bf16(a0, b0, acc[0][0]);
    acc[0][1] = wmma_bf16(a0, b1, acc[0][1]);
    acc[1][0] = wmma_bf16(a1, b0, acc[1][0]);
    acc[1][1] = wmma_bf16(a1, b1, acc[1][1]);
  }

  const int hi8 = (lane >> 4) * 8, n = lane & 15;
#pragma unroll
  for (int mt = 0; mt < 2; ++mt)
#pragma unroll
    for (int nt = 0; nt < 2; ++nt)
#pragma unroll
      for (int i = 0; i < 8; ++i) {
        int r = row0 + mt * 16 + i + hi8;          // D layout: M = i + 8*(lane>>4)
        int c = col0 + nt * 16 + n;
        wout[(size_t)r * GDIM + c] = (bf16)acc[mt][nt][i];
      }
}

// ---------------- deterministic BN stats (two-stage tree) ----------------------
__global__ __launch_bounds__(256) void stats_partial(const bf16* __restrict__ wbuf,
                                                     float* __restrict__ psum,
                                                     float* __restrict__ psq) {
  int col   = blockIdx.x * 256 + threadIdx.x;      // gridDim.x = 8 -> 2048 cols
  int chunk = blockIdx.y;                          // 64 chunks * 250 rows
  float s = 0.f, q = 0.f;
  size_t base = (size_t)chunk * 250 * GDIM + col;
  for (int r = 0; r < 250; ++r) {
    float v = (float)wbuf[base + (size_t)r * GDIM];
    s += v; q += v * v;
  }
  psum[chunk * GDIM + col] = s;
  psq [chunk * GDIM + col] = q;
}

__global__ __launch_bounds__(256) void stats_final(const float* __restrict__ psum,
                                                   const float* __restrict__ psq,
                                                   const float* __restrict__ gamma,
                                                   const float* __restrict__ beta,
                                                   float* __restrict__ scale,
                                                   float* __restrict__ shift) {
  int col = blockIdx.x * 256 + threadIdx.x;
  float s = 0.f, q = 0.f;
  for (int c = 0; c < 64; ++c) { s += psum[c * GDIM + col]; q += psq[c * GDIM + col]; }
  const float inv = 1.0f / (float)ROWS_TOT;
  float mean = s * inv;
  float var  = q * inv - mean * mean;
  float sc   = gamma[col] * rsqrtf(var + 1e-5f);
  scale[col] = sc;
  shift[col] = beta[col] - mean * sc;
}

// ---------------- init: zero h state (both buffers) + grid barrier -------------
__global__ __launch_bounds__(256) void init_state(bf16* __restrict__ hb,
                                                  unsigned* __restrict__ bar) {
  int id = blockIdx.x * 256 + threadIdx.x;
  if (id < 2 * BATCH2 * HDIM) hb[id] = (bf16)0.0f;
  if (id < 2) bar[id] = 0u;
}

// ---------------- persistent recurrence kernel ---------------------------------
// 32 WGs x 8 waves. WG wg owns h columns [wg*32, wg*32+32): gate n-tiles
// { a: wg*2, wg*2+1 ; z: 64+wg*2, 64+wg*2+1 }. Per wave: one 16x16 output tile.
// U slice lives in LDS as prepacked bf16 B-fragments (128 KB).
__global__ __launch_bounds__(256) void ligru_recur(const bf16* __restrict__ wbuf,
                                                   const Frag16* __restrict__ Upk,
                                                   const float* __restrict__ scale,
                                                   const float* __restrict__ shift,
                                                   float* __restrict__ out,
                                                   bf16* __restrict__ hb,
                                                   unsigned* __restrict__ bar) {
  __shared__ Frag16 ldsU[4][32][32];   // [local ntile][kstep][lane] = 128 KB
  __shared__ float  zx[4][32][8];      // z-gate exchange (sigmoid values)

  const int tid = threadIdx.x, wg = blockIdx.x;
  const int w = tid >> 5, lane = tid & 31;

  // Stage this WG's U fragments into LDS.
  for (int idx = tid; idx < 4 * 32 * 32; idx += 256) {
    int ntl = idx >> 10, ks = (idx >> 5) & 31, ln = idx & 31;
    int nt  = (ntl < 2) ? (wg * 2 + ntl) : (64 + wg * 2 + (ntl - 2));
    ldsU[ntl][ks][ln] = Upk[(size_t)nt * 1024 + ks * 32 + ln];
  }
  __syncthreads();

  const int p     = w & 3;             // pairs a-wave w with z-wave w+4
  const int is_z  = w >> 2;
  const int mtile = p & 1, ctile = p >> 1;
  const int m0    = mtile * 16;
  const int c0    = wg * 32 + ctile * 16;          // h-column base
  const int ntl   = is_z * 2 + ctile;
  const int gcol0 = is_z ? (HDIM + c0) : c0;
  const int n     = lane & 15;
  const int hi8   = (lane >> 4) * 8;
  const float sreg = scale[gcol0 + n];
  const float creg = shift[gcol0 + n];

  const int arow = m0 + (lane & 15);               // A-fragment source row of h
  const int akb  = (lane >> 4) << 3;

  float hreg[8] = {};                              // persistent h (a-waves only)
  bf16* h_cur = hb;
  bf16* h_nxt = hb + BATCH2 * HDIM;

  for (int t = 0; t < TSTEPS; ++t) {
    // gates tile = h @ U^T  (K = 1024 in 32 steps of 32)
    v8f acc = {};
#pragma unroll 2
    for (int ks = 0; ks < 32; ++ks) {
      const uint4* ap =
          (const uint4*)(h_cur + (size_t)arow * HDIM + ks * 32 + akb);
      Frag16 af; af.lo = ap[0]; af.hi = ap[2];     // K +0..7 and +16..23
      v16bf a = __builtin_bit_cast(v16bf, af);
      v16bf b = __builtin_bit_cast(v16bf, ldsU[ntl][ks][lane]);
      acc = wmma_bf16(a, b, acc);
    }

    // add batch-normed pre-activation (folded scale/shift)
    float g[8];
#pragma unroll
    for (int i = 0; i < 8; ++i) {
      int bm = m0 + i + hi8;                       // D layout: M = i + 8*(lane>>4)
      float wv = (float)wbuf[(size_t)(bm * TSTEPS + t) * GDIM + gcol0 + n];
      g[i] = acc[i] + wv * sreg + creg;
    }

    if (is_z) {
#pragma unroll
      for (int i = 0; i < 8; ++i)
        zx[p][lane][i] = 1.0f / (1.0f + __expf(-g[i]));
    }
    __syncthreads();

    if (!is_z) {
      // h = z*h + (1-z)*relu(a); write bf16 state + fused bidirectional output
#pragma unroll
      for (int i = 0; i < 8; ++i) {
        float z  = zx[p][lane][i];
        float at = g[i] > 0.f ? g[i] : 0.f;
        float h  = z * hreg[i] + (1.0f - z) * at;
        hreg[i]  = h;
        int bm = m0 + i + hi8;
        int hc = c0 + n;
        h_nxt[(size_t)bm * HDIM + hc] = (bf16)h;
        if (bm < 16)
          out[((size_t)(bm * TSTEPS + t)) * GDIM + hc] = h;
        else
          out[((size_t)((bm - 16) * TSTEPS + (TSTEPS - 1 - t))) * GDIM + HDIM + hc] = h;
      }
    }
    __syncthreads();

    // grid-wide barrier: release next step once all 32 WGs finished this one
    if (tid == 0) {
      __threadfence();
      unsigned ticket = atomicAdd(&bar[0], 1u);
      unsigned gen = ticket / NWG + 1u;
      if ((ticket % NWG) == (unsigned)(NWG - 1)) atomicAdd(&bar[1], 1u);
      while (__hip_atomic_load(&bar[1], __ATOMIC_ACQUIRE,
                               __HIP_MEMORY_SCOPE_AGENT) < gen) {
        __builtin_amdgcn_s_sleep(2);
      }
    }
    __syncthreads();

    bf16* tmp = h_cur; h_cur = h_nxt; h_nxt = tmp;
  }
}

// ---------------- host orchestration ------------------------------------------
extern "C" void kernel_launch(void* const* d_in, const int* in_sizes, int n_in,
                              void* d_out, int out_size, void* d_ws, size_t ws_size,
                              hipStream_t stream) {
  (void)in_sizes; (void)n_in; (void)out_size; (void)ws_size;
  const float* x  = (const float*)d_in[0];
  const float* w0 = (const float*)d_in[1];
  const float* u0 = (const float*)d_in[2];
  const float* g0 = (const float*)d_in[3];
  const float* b0 = (const float*)d_in[4];
  const float* w1 = (const float*)d_in[5];
  const float* u1 = (const float*)d_in[6];
  const float* g1 = (const float*)d_in[7];
  const float* b1 = (const float*)d_in[8];
  float* out = (float*)d_out;

  char* p = (char*)d_ws;
  auto take = [&](size_t bytes) {
    char* r = p; p += (bytes + 255) & ~(size_t)255; return r;
  };
  bf16*     wbuf = (bf16*)    take((size_t)ROWS_TOT * GDIM * sizeof(bf16)); // 65.5 MB
  Frag16*   upk  = (Frag16*)  take((size_t)128 * 32 * 32 * sizeof(Frag16)); // 4 MB
  float*    psum = (float*)   take((size_t)64 * GDIM * sizeof(float));
  float*    psq  = (float*)   take((size_t)64 * GDIM * sizeof(float));
  float*    scl  = (float*)   take(GDIM * sizeof(float));
  float*    shf  = (float*)   take(GDIM * sizeof(float));
  bf16*     hb   = (bf16*)    take((size_t)2 * BATCH2 * HDIM * sizeof(bf16));
  unsigned* bar  = (unsigned*)take(256);

  for (int layer = 0; layer < 2; ++layer) {
    const float* U  = layer ? u1 : u0;
    const float* Wm = layer ? w1 : w0;
    const float* gm = layer ? g1 : g0;
    const float* bt = layer ? b1 : b0;
    const float* in = layer ? (const float*)out : x;   // layer0 output lives in d_out
    int K = layer ? GDIM : 512;

    hipLaunchKernelGGL(pack_u,        dim3(512),     dim3(256), 0, stream, U, upk);
    hipLaunchKernelGGL(pregemm,       dim3(16, 250), dim3(256), 0, stream, in, K, Wm, wbuf);
    hipLaunchKernelGGL(stats_partial, dim3(8, 64),   dim3(256), 0, stream, wbuf, psum, psq);
    hipLaunchKernelGGL(stats_final,   dim3(8),       dim3(256), 0, stream, psum, psq, gm, bt, scl, shf);
    hipLaunchKernelGGL(init_state,    dim3(257),     dim3(256), 0, stream, hb, bar);
    hipLaunchKernelGGL(ligru_recur,   dim3(NWG),     dim3(256), 0, stream, wbuf, upk, scl, shf, out, hb, bar);
  }
}
